// EarlyExitSparseEncoderV2_48627619725805
// MI455X (gfx1250) — compile-verified, matrix operand order checked by reading, not run on hardware
//
#include <hip/hip_runtime.h>
#include <hip/hip_bf16.h>
#include <cstdint>
#include <cstddef>

// ---------------------------------------------------------------------------
// Types for CDNA5 WMMA (wave32): bf16 A/B fragments (8 VGPRs), f32 C/D (8 VGPRs)
// ---------------------------------------------------------------------------
typedef __attribute__((ext_vector_type(16))) __bf16 v16bf;
typedef __attribute__((ext_vector_type(8)))  __bf16 bf16x8;
typedef __attribute__((ext_vector_type(8)))  float  v8f;

struct ConvP {
  int Di, Hi, Wi, Cinp;        // input dims, padded input channels (mult of 8)
  int Do, Ho, Wo, Cout;        // output dims, output channels (mult of 16)
  int kd, kh, kw;
  int sd, sh, sw;
  int pd, ph, pw;
  int K, KT, Mvox;             // K = kd*kh*kw*Cinp, KT = ceil(K/32)
  int cshift;                  // log2(Cinp)
};

// ---------------------------------------------------------------------------
// Utility kernels
// ---------------------------------------------------------------------------
__global__ void zero_u32_kernel(uint32_t* p, long long n) {
  long long i = (long long)blockIdx.x * blockDim.x + threadIdx.x;
  if (i < n) p[i] = 0u;
}

// Scatter voxel features into dense channel-last bf16 grid + mask.
__global__ void scatter_kernel(const float* __restrict__ vf,
                               const int* __restrict__ coors,
                               __bf16* __restrict__ act,
                               float* __restrict__ mask,
                               int N, int Cin, int Cinp, int H, int W) {
  int i = blockIdx.x * blockDim.x + threadIdx.x;
  if (i >= N) return;
  int z = coors[i * 4 + 1];
  int y = coors[i * 4 + 2];
  int x = coors[i * 4 + 3];
  size_t vox = ((size_t)z * H + y) * W + x;
  mask[vox] = 1.0f;
  for (int c = 0; c < Cinp; ++c) {
    float v = (c < Cin) ? vf[i * Cin + c] : 0.0f;
    act[vox * Cinp + c] = (__bf16)v;
  }
}

// Repack f32 weights [Cout][Cin][kd][kh][kw] into per-lane WMMA-B fragment order:
//   elem index = ((ntile*KT + t)*32 + lane)*16 + j
//   lane<16:  n = ntile*16+lane,    chunks k = t*32 + {0..7}, {16..23}
//   lane>=16: n = ntile*16+lane-16, chunks k = t*32 + {8..15}, {24..31}
__global__ void repack_w_kernel(const float* __restrict__ w,
                                __bf16* __restrict__ wfrag,
                                int Cout, int Cin, int Cinp,
                                int kd, int kh, int kw, int K, int KT) {
  int idx = blockIdx.x * blockDim.x + threadIdx.x;
  int ntiles = Cout >> 4;
  int total = ntiles * KT * 32 * 16;
  if (idx >= total) return;
  int j     = idx & 15;
  int lane  = (idx >> 4) & 31;
  int blk   = idx >> 9;                 // ntile*KT + t
  int t     = blk % KT;
  int ntile = blk / KT;
  int hi    = lane >> 4;
  int n     = ntile * 16 + (lane & 15);
  int ch    = j >> 3;
  int kk    = t * 32 + hi * 8 + ch * 16 + (j & 7);
  float val = 0.0f;
  if (kk < K) {
    int tap = kk / Cinp;
    int c   = kk - tap * Cinp;
    if (c < Cin) {
      int khkw = kh * kw;
      int td = tap / khkw;
      int tr = tap - td * khkw;
      int th = tr / kw;
      int tw = tr - th * kw;
      val = w[((((size_t)n * Cin + c) * kd + td) * kh + th) * (size_t)kw + tw];
    }
  }
  wfrag[idx] = (__bf16)val;
}

// Mask dilation (max-pool over conv window) for non-submanifold convs.
__global__ void dilate_mask_kernel(const float* __restrict__ min_,
                                   float* __restrict__ mout,
                                   int Di, int Hi, int Wi,
                                   int Do, int Ho, int Wo,
                                   int kd, int kh, int kw,
                                   int sd, int sh, int sw,
                                   int pd, int ph, int pw) {
  int idx = blockIdx.x * blockDim.x + threadIdx.x;
  int HoWo = Ho * Wo;
  if (idx >= Do * HoWo) return;
  int zo = idx / HoWo;
  int r  = idx - zo * HoWo;
  int yo = r / Wo;
  int xo = r - yo * Wo;
  int zb = zo * sd - pd, yb = yo * sh - ph, xb = xo * sw - pw;
  float v = 0.0f;
  for (int td = 0; td < kd; ++td) {
    int zi = zb + td;
    if ((unsigned)zi >= (unsigned)Di) continue;
    for (int th = 0; th < kh; ++th) {
      int yi = yb + th;
      if ((unsigned)yi >= (unsigned)Hi) continue;
      for (int tw = 0; tw < kw; ++tw) {
        int xi = xb + tw;
        if ((unsigned)xi >= (unsigned)Wi) continue;
        float m = min_[((size_t)zi * Hi + yi) * Wi + xi];
        v = fmaxf(v, m);
      }
    }
  }
  mout[idx] = v;
}

// Count of active voxels (BN denominator).
__global__ void mask_count_kernel(const float* __restrict__ mask, int n, float* cnt) {
  __shared__ float s;
  if (threadIdx.x == 0) s = 0.0f;
  __syncthreads();
  float v = 0.0f;
  for (int i = blockIdx.x * blockDim.x + threadIdx.x; i < n; i += gridDim.x * blockDim.x)
    v += mask[i];
  atomicAdd(&s, v);
  __syncthreads();
  if (threadIdx.x == 0) atomicAdd(cnt, s);
}

// ---------------------------------------------------------------------------
// Shared epilogue: masked writeback + BN sum/sumsq via LDS + global atomics.
// ---------------------------------------------------------------------------
template <int NT>
__device__ __forceinline__
void conv_epilogue(const v8f (&acc)[NT], bool active, int mtile, int lane,
                   const float* __restrict__ maskOut, __bf16* __restrict__ y,
                   float* __restrict__ gsum, float* __restrict__ gssq,
                   float* s_sum, float* s_ssq, const ConvP& p) {
  float lsum[NT], lssq[NT];
#pragma unroll
  for (int j = 0; j < NT; ++j) { lsum[j] = 0.0f; lssq[j] = 0.0f; }
  if (active) {
    const int ncol  = lane & 15;
    const int rbase = mtile * 16 + 8 * (lane >> 4);
#pragma unroll
    for (int r = 0; r < 8; ++r) {
      int m = rbase + r;
      if (m < p.Mvox) {
        float mv = maskOut[m];
        size_t rowoff = (size_t)m * p.Cout;
#pragma unroll
        for (int j = 0; j < NT; ++j) {
          float v = acc[j][r] * mv;
          y[rowoff + j * 16 + ncol] = (__bf16)v;
          lsum[j] += v;
          lssq[j] += v * v;
        }
      }
    }
  }
#pragma unroll
  for (int j = 0; j < NT; ++j) {
    atomicAdd(&s_sum[j * 16 + (lane & 15)], lsum[j]);   // ds_add_f32
    atomicAdd(&s_ssq[j * 16 + (lane & 15)], lssq[j]);
  }
  __syncthreads();
  for (int c = threadIdx.x; c < NT * 16; c += 128) {
    atomicAdd(&gsum[c], s_sum[c]);
    atomicAdd(&gssq[c], s_ssq[c]);
  }
}

// ---------------------------------------------------------------------------
// Generic implicit-GEMM conv (3x3x3 taps hardcoded) for Cinp in {8,16}.
// One wave = 16 voxels x Cout; A fragment gathered per K-tile, reused NT times.
// ---------------------------------------------------------------------------
template <int NT>
__global__ __launch_bounds__(128)
void conv_wmma_generic(const __bf16* __restrict__ act,
                       const __bf16* __restrict__ wfrag,
                       const float* __restrict__ maskOut,
                       __bf16* __restrict__ y,
                       float* __restrict__ gsum, float* __restrict__ gssq,
                       ConvP p) {
  __shared__ float s_sum[NT * 16];
  __shared__ float s_ssq[NT * 16];
  const int lane = threadIdx.x & 31;
  const int wave = threadIdx.x >> 5;
  for (int c = threadIdx.x; c < NT * 16; c += 128) { s_sum[c] = 0.0f; s_ssq[c] = 0.0f; }
  __syncthreads();

  const int mtile = blockIdx.x * 4 + wave;
  const int HoWo  = p.Ho * p.Wo;
  const bool active = (mtile * 16 < p.Mvox);          // wave-uniform

  const v8f vz = {};
  v8f acc[NT];
#pragma unroll
  for (int j = 0; j < NT; ++j) acc[j] = vz;

  if (active) {
    const int hi   = lane >> 4;
    const int mrow = lane & 15;
    const int m    = mtile * 16 + mrow;
    const bool mvalid = (m < p.Mvox);
    const int mm = mvalid ? m : (p.Mvox - 1);
    int zo = mm / HoWo;
    int rr = mm - zo * HoWo;
    int yo = rr / p.Wo;
    int xo = rr - yo * p.Wo;
    const int zb = zo * p.sd - p.pd;
    const int yb = yo * p.sh - p.ph;
    const int xb = xo * p.sw - p.pw;

    for (int t = 0; t < p.KT; ++t) {
      v16bf a;
#pragma unroll
      for (int ch = 0; ch < 2; ++ch) {
        bf16x8 frag;
#pragma unroll
        for (int q = 0; q < 8; ++q) frag[q] = (__bf16)0.0f;
        int kstart = t * 32 + hi * 8 + ch * 16;
        if (mvalid && kstart < p.K) {
          int tap = kstart >> p.cshift;                // Cinp power of two
          int c0  = kstart - (tap << p.cshift);
          int td  = tap / 9;                           // 3x3x3: compile-time magic div
          int tr  = tap - td * 9;
          int th  = tr / 3;
          int tw  = tr - th * 3;
          int zi = zb + td, yi = yb + th, xi = xb + tw;
          if ((unsigned)zi < (unsigned)p.Di &&
              (unsigned)yi < (unsigned)p.Hi &&
              (unsigned)xi < (unsigned)p.Wi) {
            size_t base = (((size_t)zi * p.Hi + yi) * p.Wi + xi) * (size_t)p.Cinp + c0;
            frag = *reinterpret_cast<const bf16x8*>(act + base);   // global_load_b128
          }
        }
#pragma unroll
        for (int q = 0; q < 8; ++q) a[ch * 8 + q] = frag[q];
      }
#pragma unroll
      for (int j = 0; j < NT; ++j) {
        const bf16x8* wp = reinterpret_cast<const bf16x8*>(
            wfrag + (((size_t)j * p.KT + t) * 512) + (size_t)lane * 16);
        bf16x8 b0 = wp[0];
        bf16x8 b1 = wp[1];
        v16bf b;
#pragma unroll
        for (int q = 0; q < 8; ++q) { b[q] = b0[q]; b[8 + q] = b1[q]; }
        acc[j] = __builtin_amdgcn_wmma_f32_16x16x32_bf16(
            false, a, false, b, (short)0, acc[j], false, false);
      }
    }
  }
  conv_epilogue<NT>(acc, active, mtile, lane, maskOut, y, gsum, gssq, s_sum, s_ssq, p);
}

// ---------------------------------------------------------------------------
// Tap-outer implicit-GEMM conv for Cinp == CPT*32 (CPT = 1 or 2).
// Incremental tap counters: no divisions; one bounds check + base address per
// tap; CPT K-tiles per tap are pure address increments. A reused for NT WMMAs.
// ---------------------------------------------------------------------------
template <int NT, int CPT>
__global__ __launch_bounds__(128)
void conv_wmma_tap(const __bf16* __restrict__ act,
                   const __bf16* __restrict__ wfrag,
                   const float* __restrict__ maskOut,
                   __bf16* __restrict__ y,
                   float* __restrict__ gsum, float* __restrict__ gssq,
                   ConvP p) {
  __shared__ float s_sum[NT * 16];
  __shared__ float s_ssq[NT * 16];
  const int lane = threadIdx.x & 31;
  const int wave = threadIdx.x >> 5;
  for (int c = threadIdx.x; c < NT * 16; c += 128) { s_sum[c] = 0.0f; s_ssq[c] = 0.0f; }
  __syncthreads();

  const int mtile = blockIdx.x * 4 + wave;
  const int HoWo  = p.Ho * p.Wo;
  const bool active = (mtile * 16 < p.Mvox);          // wave-uniform

  const v8f vz = {};
  v8f acc[NT];
#pragma unroll
  for (int j = 0; j < NT; ++j) acc[j] = vz;

  if (active) {
    const int hi   = lane >> 4;
    const int mrow = lane & 15;
    const int m    = mtile * 16 + mrow;
    const bool mvalid = (m < p.Mvox);
    const int mm = mvalid ? m : (p.Mvox - 1);
    int zo = mm / HoWo;
    int rr = mm - zo * HoWo;
    int yo = rr / p.Wo;
    int xo = rr - yo * p.Wo;
    const int zb = zo * p.sd - p.pd;
    const int yb = yo * p.sh - p.ph;
    const int xb = xo * p.sw - p.pw;
    const int coff = hi * 8;                          // half-wave k offset
    const int ntaps = p.kd * p.kh * p.kw;

    int td = 0, th = 0, tw = 0;
    for (int tap = 0; tap < ntaps; ++tap) {
      const int zi = zb + td, yi = yb + th, xi = xb + tw;
      const bool inb = mvalid &&
                       (unsigned)zi < (unsigned)p.Di &&
                       (unsigned)yi < (unsigned)p.Hi &&
                       (unsigned)xi < (unsigned)p.Wi;
      const __bf16* src =
          act + (((size_t)zi * p.Hi + yi) * p.Wi + xi) * (size_t)(CPT * 32) + coff;
#pragma unroll
      for (int ct = 0; ct < CPT; ++ct) {
        v16bf a;
#pragma unroll
        for (int ch = 0; ch < 2; ++ch) {
          bf16x8 frag;
#pragma unroll
          for (int q = 0; q < 8; ++q) frag[q] = (__bf16)0.0f;
          if (inb)
            frag = *reinterpret_cast<const bf16x8*>(src + ct * 32 + ch * 16);  // b128
#pragma unroll
          for (int q = 0; q < 8; ++q) a[ch * 8 + q] = frag[q];
        }
        const int t = tap * CPT + ct;
#pragma unroll
        for (int j = 0; j < NT; ++j) {
          const bf16x8* wp = reinterpret_cast<const bf16x8*>(
              wfrag + (((size_t)j * p.KT + t) * 512) + (size_t)lane * 16);
          bf16x8 b0 = wp[0];
          bf16x8 b1 = wp[1];
          v16bf b;
#pragma unroll
          for (int q = 0; q < 8; ++q) { b[q] = b0[q]; b[8 + q] = b1[q]; }
          acc[j] = __builtin_amdgcn_wmma_f32_16x16x32_bf16(
              false, a, false, b, (short)0, acc[j], false, false);
        }
      }
      // incremental tap counters (no division)
      if (++tw == p.kw) { tw = 0; if (++th == p.kh) { th = 0; ++td; } }
    }
  }
  conv_epilogue<NT>(acc, active, mtile, lane, maskOut, y, gsum, gssq, s_sum, s_ssq, p);
}

// ---------------------------------------------------------------------------
// BN finalize: scale = gamma*rsqrt(var+eps), shift = beta - mean*scale.
// ---------------------------------------------------------------------------
__global__ void bn_finalize_kernel(const float* __restrict__ sum,
                                   const float* __restrict__ ssq,
                                   const float* __restrict__ cnt,
                                   const float* __restrict__ gamma,
                                   const float* __restrict__ beta,
                                   float* __restrict__ scale,
                                   float* __restrict__ shift, int C) {
  int c = blockIdx.x * blockDim.x + threadIdx.x;
  if (c >= C) return;
  float n    = fmaxf(cnt[0], 1.0f);
  float mean = sum[c] / n;
  float var  = ssq[c] / n - mean * mean;
  float sc   = gamma[c] * rsqrtf(var + 1.0e-3f);
  scale[c] = sc;
  shift[c] = beta[c] - mean * sc;
}

// BN + ReLU + mask, in place on the bf16 ping-pong buffer.
__global__ void bn_apply_kernel(__bf16* __restrict__ y,
                                const float* __restrict__ mask,
                                const float* __restrict__ scale,
                                const float* __restrict__ shift,
                                int Mvox, int C) {
  long long idx = (long long)blockIdx.x * blockDim.x + threadIdx.x;
  if (idx >= (long long)Mvox * C) return;
  int m = (int)(idx / C);
  int c = (int)(idx - (long long)m * C);
  float v = (float)y[idx] * scale[c] + shift[c];
  v = fmaxf(v, 0.0f) * mask[m];
  y[idx] = (__bf16)v;
}

// Final layer: BN + ReLU + mask, written f32 to d_out with (C,D)->C*D reshape.
__global__ void bn_apply_out_kernel(const __bf16* __restrict__ y,
                                    const float* __restrict__ mask,
                                    const float* __restrict__ scale,
                                    const float* __restrict__ shift,
                                    float* __restrict__ out,
                                    int Do, int Ho, int Wo, int C) {
  long long idx = (long long)blockIdx.x * blockDim.x + threadIdx.x;
  long long total = (long long)Do * Ho * Wo * C;
  if (idx >= total) return;
  int m = (int)(idx / C);
  int c = (int)(idx - (long long)m * C);
  int z = m / (Ho * Wo);
  int r = m - z * Ho * Wo;
  int yy = r / Wo;
  int xx = r - yy * Wo;
  float v = (float)y[idx] * scale[c] + shift[c];
  v = fmaxf(v, 0.0f) * mask[m];
  out[(((size_t)c * Do + z) * Ho + yy) * Wo + xx] = v;
}

// ---------------------------------------------------------------------------
// Host orchestration
// ---------------------------------------------------------------------------
static inline size_t align256(size_t x) { return (x + 255) & ~(size_t)255; }
static inline int cdiv(int a, int b) { return (a + b - 1) / b; }

extern "C" void kernel_launch(void* const* d_in, const int* in_sizes, int n_in,
                              void* d_out, int out_size, void* d_ws, size_t ws_size,
                              hipStream_t stream) {
  const float* vf    = (const float*)d_in[0];
  const int*   coors = (const int*)d_in[1];
  const int Nvox = in_sizes[0] / 5;

  struct L { int Cin, Cout, kd, kh, kw, sd, sh, sw, pd, ph, pw, w; bool subm; };
  const L Ls[11] = {
    { 5,  16, 3,3,3, 1,1,1, 1,1,1,  3, true  },   // w_in
    {16,  16, 3,3,3, 1,1,1, 1,1,1,  6, true  },   // w1a
    {16,  32, 3,3,3, 2,2,2, 1,1,1,  9, false },   // w2a
    {32,  32, 3,3,3, 1,1,1, 1,1,1, 12, true  },   // w2b
    {32,  32, 3,3,3, 1,1,1, 1,1,1, 15, true  },   // w2c
    {32,  64, 3,3,3, 2,2,2, 1,1,1, 18, false },   // w3a
    {64,  64, 3,3,3, 1,1,1, 1,1,1, 21, true  },   // w3b
    {64,  64, 3,3,3, 1,1,1, 1,1,1, 24, true  },   // w3c
    {64,  64, 3,3,3, 1,1,1, 1,1,1, 27, false },   // wde
    {64,  64, 1,1,1, 1,1,1, 0,0,0, 30, false },   // wdl (identity dilation)
    {64, 128, 3,1,1, 2,1,1, 0,0,0, 33, false },   // wo
  };

  const int D0 = 41, H0 = 256, W0 = 224;
  const size_t R0 = (size_t)D0 * H0 * W0;

  // Workspace layout (inter-layer working set stays inside the 192MB L2).
  char* ws = (char*)d_ws;
  size_t off = 0;
  const size_t actBytes = align256(R0 * 16 * sizeof(__bf16));   // worst case: R0 x 16ch
  __bf16* actA = (__bf16*)(ws + off); off += actBytes;
  __bf16* actB = (__bf16*)(ws + off); off += actBytes;
  float* maskA = (float*)(ws + off);  off += align256(R0 * sizeof(float));
  float* maskB = (float*)(ws + off);  off += align256(R0 * sizeof(float));
  float* stats = (float*)(ws + off);  off += align256(520 * sizeof(float));
  float* gsum  = stats;          // [128]
  float* gssq  = stats + 128;    // [128]
  float* cnt   = stats + 256;    // [1] (+pad)
  float* scale = stats + 260;    // [128]
  float* shift = stats + 388;    // [128]
  __bf16* wfrag[11];
  int KTs[11], Cinps[11];
  for (int i = 0; i < 11; ++i) {
    const L& l = Ls[i];
    int Cinp = (i == 0) ? 8 : l.Cin;
    int K  = l.kd * l.kh * l.kw * Cinp;
    int KT = (K + 31) / 32;
    Cinps[i] = Cinp; KTs[i] = KT;
    size_t elems = (size_t)(l.Cout / 16) * KT * 32 * 16;
    wfrag[i] = (__bf16*)(ws + off);
    off += align256(elems * sizeof(__bf16));
  }
  (void)ws_size;

  // --- Init: zero the layer-0 activation region + full-res mask, scatter. ---
  {
    long long nz = (long long)(R0 * 8 * sizeof(__bf16) / 4);
    zero_u32_kernel<<<dim3((unsigned)((nz + 255) / 256)), dim3(256), 0, stream>>>((uint32_t*)actA, nz);
    long long nm = (long long)R0;
    zero_u32_kernel<<<dim3((unsigned)((nm + 255) / 256)), dim3(256), 0, stream>>>((uint32_t*)maskA, nm);
    scatter_kernel<<<dim3(cdiv(Nvox, 256)), dim3(256), 0, stream>>>(
        vf, coors, actA, maskA, Nvox, 5, 8, H0, W0);
  }

  // --- Repack all weights to WMMA fragment order (bf16). ---
  for (int i = 0; i < 11; ++i) {
    const L& l = Ls[i];
    int K = l.kd * l.kh * l.kw * Cinps[i];
    int total = (l.Cout / 16) * KTs[i] * 32 * 16;
    repack_w_kernel<<<dim3(cdiv(total, 256)), dim3(256), 0, stream>>>(
        (const float*)d_in[l.w], wfrag[i], l.Cout, l.Cin, Cinps[i],
        l.kd, l.kh, l.kw, K, KTs[i]);
  }

  // --- Layer loop ---
  int Di = D0, Hi = H0, Wi = W0;
  __bf16* bIn  = actA;
  __bf16* bOut = actB;
  float* mCur = maskA;
  float* mAlt = maskB;

  for (int i = 0; i < 11; ++i) {
    const L& l = Ls[i];
    int Do = (Di + 2 * l.pd - l.kd) / l.sd + 1;
    int Ho = (Hi + 2 * l.ph - l.kh) / l.sh + 1;
    int Wo = (Wi + 2 * l.pw - l.kw) / l.sw + 1;
    int Mvox = Do * Ho * Wo;

    if (!l.subm) {
      dilate_mask_kernel<<<dim3(cdiv(Mvox, 256)), dim3(256), 0, stream>>>(
          mCur, mAlt, Di, Hi, Wi, Do, Ho, Wo,
          l.kd, l.kh, l.kw, l.sd, l.sh, l.sw, l.pd, l.ph, l.pw);
      float* t = mCur; mCur = mAlt; mAlt = t;
    }

    // zero stats (sum/ssq/cnt), then count active voxels
    zero_u32_kernel<<<dim3(2), dim3(256), 0, stream>>>((uint32_t*)stats, 260);
    mask_count_kernel<<<dim3(256), dim3(256), 0, stream>>>(mCur, Mvox, cnt);

    // conv (implicit GEMM, WMMA) — one wave covers 16 voxels x full Cout
    ConvP p;
    p.Di = Di; p.Hi = Hi; p.Wi = Wi; p.Cinp = Cinps[i];
    p.Do = Do; p.Ho = Ho; p.Wo = Wo; p.Cout = l.Cout;
    p.kd = l.kd; p.kh = l.kh; p.kw = l.kw;
    p.sd = l.sd; p.sh = l.sh; p.sw = l.sw;
    p.pd = l.pd; p.ph = l.ph; p.pw = l.pw;
    p.K = l.kd * l.kh * l.kw * p.Cinp; p.KT = KTs[i]; p.Mvox = Mvox;
    p.cshift = (p.Cinp == 8) ? 3 : (p.Cinp == 16) ? 4 : (p.Cinp == 32) ? 5 : 6;
    int mtiles = cdiv(Mvox, 16);
    dim3 grid(cdiv(mtiles, 4));
    const dim3 blk(128);
    const int nt  = l.Cout / 16;
    const int cpt = p.Cinp >> 5;          // K-tiles per tap (0 => generic path)
    if (cpt == 0) {
      if (nt == 1)
        conv_wmma_generic<1><<<grid, blk, 0, stream>>>(bIn, wfrag[i], mCur, bOut, gsum, gssq, p);
      else
        conv_wmma_generic<2><<<grid, blk, 0, stream>>>(bIn, wfrag[i], mCur, bOut, gsum, gssq, p);
    } else if (cpt == 1) {
      if (nt == 2)
        conv_wmma_tap<2, 1><<<grid, blk, 0, stream>>>(bIn, wfrag[i], mCur, bOut, gsum, gssq, p);
      else
        conv_wmma_tap<4, 1><<<grid, blk, 0, stream>>>(bIn, wfrag[i], mCur, bOut, gsum, gssq, p);
    } else {
      if (nt == 4)
        conv_wmma_tap<4, 2><<<grid, blk, 0, stream>>>(bIn, wfrag[i], mCur, bOut, gsum, gssq, p);
      else
        conv_wmma_tap<8, 2><<<grid, blk, 0, stream>>>(bIn, wfrag[i], mCur, bOut, gsum, gssq, p);
    }

    // BN finalize
    bn_finalize_kernel<<<dim3(1), dim3(128), 0, stream>>>(
        gsum, gssq, cnt, (const float*)d_in[l.w + 1], (const float*)d_in[l.w + 2],
        scale, shift, l.Cout);

    // BN + ReLU + mask
    long long total = (long long)Mvox * l.Cout;
    if (i == 10) {
      bn_apply_out_kernel<<<dim3((unsigned)((total + 255) / 256)), dim3(256), 0, stream>>>(
          bOut, mCur, scale, shift, (float*)d_out, Do, Ho, Wo, l.Cout);
    } else {
      bn_apply_kernel<<<dim3((unsigned)((total + 255) / 256)), dim3(256), 0, stream>>>(
          bOut, mCur, scale, shift, Mvox, l.Cout);
    }

    // ping-pong
    __bf16* t = bIn; bIn = bOut; bOut = t;
    Di = Do; Hi = Ho; Wi = Wo;
  }
  (void)out_size; (void)n_in;
}
